// MultiLayerGCN_49520972923234
// MI455X (gfx1250) — matrix operand order repaired and checked by reading, not compiled, not run on hardware
//
#include <hip/hip_runtime.h>
#include <hip/hip_bf16.h>
#include <math.h>

// MI455X / gfx1250: wave32, WMMA 16x16x32 f16 -> f32 accum.
typedef __attribute__((ext_vector_type(16))) _Float16 v16h;
typedef __attribute__((ext_vector_type(8)))  _Float16 v8h;
typedef __attribute__((ext_vector_type(8)))  float    v8f;

#define N_NODES 50000
#define N_EDGES 800000
#define BN_EPS  1e-5f

__device__ __forceinline__ float gelu_exact(float x) {
    return 0.5f * x * (1.0f + erff(x * 0.70710678118654752440f));
}

// ---------------------------------------------------------------------------
// Tiled WMMA GEMM: C[M,N] = A[M,K] @ W[K,N] (+bias). A,W fp32 row-major,
// converted to f16 in LDS. Block = 256 threads = 8 waves; block tile 64x64;
// each wave owns a 16x32 C strip (two 16x16 tiles sharing the A fragment);
// K stepped by 64 -> 4 WMMAs per stage on 4 INDEPENDENT accumulators (no
// WMMA->WMMA RAW hazard; chunks summed in the epilogue).
//   - K % 16 == 0 and N % 16 == 0 for every layer, so staging guards are
//     chunk-uniform (no per-element divergence).
//   - sA row stride 72 halves (144B): every 8-half fragment run is 16B-aligned
//     -> ds_load_b128. sB is stored TRANSPOSED [n][k], stride 80 halves
//     (160B): a full B fragment is 16 contiguous halves, 32B-aligned.
// ---------------------------------------------------------------------------
#define SA_STRIDE 72
#define SB_STRIDE 80

__global__ void __launch_bounds__(256)
gemm_wmma_f16(const float* __restrict__ A, const float* __restrict__ W,
              const float* __restrict__ bias, float* __restrict__ C,
              int M, int K, int N)
{
    __shared__ _Float16 sA[64 * SA_STRIDE];   // 64 rows x 64 k
    __shared__ _Float16 sBt[64 * SB_STRIDE];  // 64 n   x 64 k (transposed)

    const int tid    = threadIdx.x;
    const int lane   = tid & 31;
    const int waveId = tid >> 5;
    const int wm     = waveId & 3;     // row sub-tile 0..3 (16 rows each)
    const int wn     = waveId >> 2;    // col strip 0..1 (32 cols each)
    const int rowBase = blockIdx.x * 64;
    const int colBase = blockIdx.y * 64;
    const int half = lane >> 4;        // K-half selector
    const int l16  = lane & 15;

    v8f acc[2][2] = {};                // [n-tile][k-chunk]

    const _Float16* fragArow  = &sA [(wm * 16 + l16) * SA_STRIDE];
    const _Float16* fragBrow0 = &sBt[(wn * 32 + l16) * SB_STRIDE];
    const _Float16* fragBrow1 = &sBt[(wn * 32 + 16 + l16) * SB_STRIDE];

    for (int kt = 0; kt < K; kt += 64) {
        // ---- stage A: 64 rows x 64 k; each thread 16 contiguous k ----
        {
            const int r  = tid >> 2;             // 0..63
            const int k0 = (tid & 3) << 4;       // 0,16,32,48
            const int gr = rowBase + r;
            v8h h0 = {}, h1 = {};
            if (gr < M && (kt + k0) < K) {       // chunk-uniform (K % 16 == 0)
                const float4* p = (const float4*)&A[(long)gr * K + kt + k0];
                float4 f0 = p[0], f1 = p[1], f2 = p[2], f3 = p[3];
                h0[0]=(_Float16)f0.x; h0[1]=(_Float16)f0.y; h0[2]=(_Float16)f0.z; h0[3]=(_Float16)f0.w;
                h0[4]=(_Float16)f1.x; h0[5]=(_Float16)f1.y; h0[6]=(_Float16)f1.z; h0[7]=(_Float16)f1.w;
                h1[0]=(_Float16)f2.x; h1[1]=(_Float16)f2.y; h1[2]=(_Float16)f2.z; h1[3]=(_Float16)f2.w;
                h1[4]=(_Float16)f3.x; h1[5]=(_Float16)f3.y; h1[6]=(_Float16)f3.z; h1[7]=(_Float16)f3.w;
            }
            *(v8h*)&sA[r * SA_STRIDE + k0]     = h0;   // 16B-aligned -> b128
            *(v8h*)&sA[r * SA_STRIDE + k0 + 8] = h1;
        }
        // ---- stage B transposed: 64 k x 64 n; each thread 16 n-contiguous ----
        {
            const int kr = tid >> 2;             // 0..63
            const int n0 = (tid & 3) << 4;       // 0,16,32,48
            _Float16 hv[16];
            if ((kt + kr) < K && (colBase + n0) < N) {  // chunk-uniform (N % 16 == 0)
                const float4* p = (const float4*)&W[(long)(kt + kr) * N + colBase + n0];
                float4 f0 = p[0], f1 = p[1], f2 = p[2], f3 = p[3];
                hv[0] =(_Float16)f0.x; hv[1] =(_Float16)f0.y; hv[2] =(_Float16)f0.z; hv[3] =(_Float16)f0.w;
                hv[4] =(_Float16)f1.x; hv[5] =(_Float16)f1.y; hv[6] =(_Float16)f1.z; hv[7] =(_Float16)f1.w;
                hv[8] =(_Float16)f2.x; hv[9] =(_Float16)f2.y; hv[10]=(_Float16)f2.z; hv[11]=(_Float16)f2.w;
                hv[12]=(_Float16)f3.x; hv[13]=(_Float16)f3.y; hv[14]=(_Float16)f3.z; hv[15]=(_Float16)f3.w;
            } else {
                #pragma unroll
                for (int e = 0; e < 16; ++e) hv[e] = (_Float16)0.0f;
            }
            #pragma unroll
            for (int e = 0; e < 16; ++e)
                sBt[(n0 + e) * SB_STRIDE + kr] = hv[e];  // transposed write
        }
        __syncthreads();

        // ---- 4 WMMAs per stage: 2 k-chunks x 2 n-tiles, A fragment reused ----
        #pragma unroll
        for (int c = 0; c < 2; ++c) {
            // A fragment (16x32): elem e -> klocal = half*16 + e (contiguous)
            v8h lo = *(const v8h*)&fragArow[c * 32 + half * 16];
            v8h hi = *(const v8h*)&fragArow[c * 32 + half * 16 + 8];
            v16h af = __builtin_shufflevector(lo, hi,
                        0,1,2,3,4,5,6,7, 8,9,10,11,12,13,14,15);
            // B fragments (32x16): elem e -> klocal = half*16 + e (contiguous)
            v16h bf0 = *(const v16h*)&fragBrow0[c * 32 + half * 16];
            v16h bf1 = *(const v16h*)&fragBrow1[c * 32 + half * 16];
            acc[0][c] = __builtin_amdgcn_wmma_f32_16x16x32_f16(
                            false, af, false, bf0, (short)0, acc[0][c], false, false);
            acc[1][c] = __builtin_amdgcn_wmma_f32_16x16x32_f16(
                            false, af, false, bf1, (short)0, acc[1][c], false, false);
        }
        __syncthreads();
    }

    // C layout: VGPR r -> M = 8*half + r, N = lane&15
    #pragma unroll
    for (int nt = 0; nt < 2; ++nt) {
        const int n = colBase + wn * 32 + nt * 16 + l16;
        if (n < N) {
            const float bv = bias ? bias[n] : 0.0f;
            v8f a = acc[nt][0] + acc[nt][1];
            #pragma unroll
            for (int r = 0; r < 8; ++r) {
                const int m = rowBase + wm * 16 + half * 8 + r;
                if (m < M) C[(long)m * N + n] = a[r] + bv;
            }
        }
    }
}

// ---------------------------------------------------------------------------
// Degree / normalization
// ---------------------------------------------------------------------------
__global__ void deg_init(float* deg, int n) {
    int i = blockIdx.x * blockDim.x + threadIdx.x;
    if (i < n) deg[i] = 1.0f;                       // self-loop contribution
}
__global__ void deg_accum(const int* __restrict__ dst, float* deg, int e) {
    int i = blockIdx.x * blockDim.x + threadIdx.x;
    if (i < e) atomicAdd(&deg[dst[i]], 1.0f);
}
__global__ void deg_rsqrt(float* deg, int n) {
    int i = blockIdx.x * blockDim.x + threadIdx.x;
    if (i < n) { float d = deg[i]; deg[i] = (d > 0.0f) ? rsqrtf(d) : 0.0f; }
}

// agg = bias + self-loop term (dinv[i]^2 * hw[i]);  d = 1<<dshift
__global__ void agg_init(const float* __restrict__ hw, const float* __restrict__ dinv,
                         const float* __restrict__ bias, float* __restrict__ agg,
                         int n, int dshift)
{
    long t = (long)blockIdx.x * blockDim.x + threadIdx.x;
    if (t < ((long)n << dshift)) {
        int i = (int)(t >> dshift), j = (int)(t & ((1 << dshift) - 1));
        float w = dinv[i];
        agg[t] = bias[j] + hw[t] * w * w;
    }
}

// (1<<lshift) lanes per edge: agg[dst] += hw[src] * dinv[src]*dinv[dst]
__global__ void __launch_bounds__(256)
edge_scatter(const float* __restrict__ hw, const int* __restrict__ src,
             const int* __restrict__ dst, const float* __restrict__ dinv,
             float* __restrict__ agg, int e, int d, int lshift)
{
    long gid = (long)blockIdx.x * blockDim.x + threadIdx.x;
    int  ei   = (int)(gid >> lshift);
    int  lane = (int)(gid & ((1 << lshift) - 1));
    if (ei < e) {
        // cover the random row gather: prefetch a later edge's source row
        if (lane == 0) {
            int ej = ei + 64;
            if (ej < e) __builtin_prefetch(&hw[(long)src[ej] * d], 0, 1);
        }
        int s = src[ei], t = dst[ei];
        float w = dinv[s] * dinv[t];
        for (int j = lane; j < d; j += (1 << lshift))
            atomicAdd(&agg[(long)t * d + j], hw[(long)s * d + j] * w);
    }
}

// ---------------------------------------------------------------------------
// BatchNorm batch statistics (column sums); 256 % d == 0 for all d used,
// so each thread owns a fixed column -> register accumulate, one atomic.
// ---------------------------------------------------------------------------
__global__ void bn_zero(float* sums, float* sumsq, int d) {
    int j = blockIdx.x * blockDim.x + threadIdx.x;
    if (j < d) { sums[j] = 0.0f; sumsq[j] = 0.0f; }
}
__global__ void __launch_bounds__(256)
bn_partial(const float* __restrict__ Z, float* sums, float* sumsq,
           int n, int d, int rowsPerBlock)
{
    long r0 = (long)blockIdx.x * rowsPerBlock;
    long r1 = r0 + rowsPerBlock; if (r1 > n) r1 = n;
    long idx = r0 * d + threadIdx.x;
    long end = r1 * d;
    int  j = threadIdx.x & (d - 1);   // constant per thread (d power of two)
    float s = 0.0f, q = 0.0f;
    for (; idx < end; idx += 256) { float v = Z[idx]; s += v; q += v * v; }
    atomicAdd(&sums[j], s);
    atomicAdd(&sumsq[j], q);
}
__global__ void bn_finalize(float* sums, float* sumsq, int n, int d) {
    int j = blockIdx.x * blockDim.x + threadIdx.x;
    if (j < d) {
        float m = sums[j] / (float)n;
        float v = sumsq[j] / (float)n - m * m;
        sums[j]  = m;
        sumsq[j] = rsqrtf(fmaxf(v, 0.0f) + BN_EPS);
    }
}

// ---------------------------------------------------------------------------
// Fused BN-apply + GELU (+ optional LayerNorm). One wave32 per row; row kept
// in registers (d<=256 -> <=8 values/lane); LN stats via shfl_xor reduction.
// ---------------------------------------------------------------------------
__global__ void __launch_bounds__(256)
bn_gelu_ln(const float* __restrict__ Z, const float* __restrict__ mean,
           const float* __restrict__ rstd, const float* __restrict__ bng,
           const float* __restrict__ bnb, const float* __restrict__ lng,
           const float* __restrict__ lnb, float* __restrict__ out,
           int n, int d, int do_ln)
{
    const int lane = threadIdx.x & 31;
    const int row  = blockIdx.x * 8 + (threadIdx.x >> 5);
    if (row >= n) return;

    float y[8];
    float s = 0.0f, q = 0.0f;
    int c = 0;
    for (int j = lane; j < d; j += 32) {
        float v = Z[(long)row * d + j];
        v = (v - mean[j]) * rstd[j] * bng[j] + bnb[j];
        v = gelu_exact(v);
        y[c++] = v;
        s += v; q += v * v;
    }
    if (!do_ln) {
        c = 0;
        for (int j = lane; j < d; j += 32) out[(long)row * d + j] = y[c++];
        return;
    }
    #pragma unroll
    for (int m = 16; m >= 1; m >>= 1) {
        s += __shfl_xor(s, m, 32);
        q += __shfl_xor(q, m, 32);
    }
    float rm = s / (float)d;
    float rv = q / (float)d - rm * rm;
    float ri = rsqrtf(fmaxf(rv, 0.0f) + BN_EPS);
    c = 0;
    for (int j = lane; j < d; j += 32)
        out[(long)row * d + j] = (y[c++] - rm) * ri * lng[j] + lnb[j];
}

// ---------------------------------------------------------------------------
// Prediction MLP 16->32->16->8->1 (LN+GELU, LN+GELU, GELU, linear).
// One thread per node; weights staged in LDS (~5 KB).
// ---------------------------------------------------------------------------
__global__ void __launch_bounds__(256)
pred_mlp(const float* __restrict__ H,
         const float* W1, const float* b1, const float* g1, const float* be1,
         const float* W2, const float* b2, const float* g2, const float* be2,
         const float* W3, const float* b3, const float* W4, const float* b4,
         float* __restrict__ out, int n)
{
    __shared__ float sW1[512], sb1[32], sg1[32], sbe1[32];
    __shared__ float sW2[512], sb2[16], sg2[16], sbe2[16];
    __shared__ float sW3[128], sb3[8], sW4[8], sb4;
    const int tid = threadIdx.x;
    for (int i = tid; i < 512; i += 256) { sW1[i] = W1[i]; sW2[i] = W2[i]; }
    if (tid < 128) sW3[tid] = W3[tid];
    if (tid < 32) { sb1[tid] = b1[tid]; sg1[tid] = g1[tid]; sbe1[tid] = be1[tid]; }
    if (tid < 16) { sb2[tid] = b2[tid]; sg2[tid] = g2[tid]; sbe2[tid] = be2[tid]; }
    if (tid < 8)  { sb3[tid] = b3[tid]; sW4[tid] = W4[tid]; }
    if (tid == 0) sb4 = b4[0];
    __syncthreads();

    const int i = blockIdx.x * 256 + tid;
    if (i >= n) return;

    float h[16];
    #pragma unroll
    for (int j = 0; j < 16; ++j) h[j] = H[(long)i * 16 + j];

    // 16 -> 32, LN, GELU
    float p[32];
    float m = 0.0f;
    #pragma unroll
    for (int o = 0; o < 32; ++o) {
        float a = sb1[o];
        #pragma unroll
        for (int j = 0; j < 16; ++j) a += h[j] * sW1[j * 32 + o];
        p[o] = a; m += a;
    }
    m *= (1.0f / 32.0f);
    float v = 0.0f;
    #pragma unroll
    for (int o = 0; o < 32; ++o) { float t = p[o] - m; v += t * t; }
    float ri = rsqrtf(v * (1.0f / 32.0f) + BN_EPS);
    #pragma unroll
    for (int o = 0; o < 32; ++o)
        p[o] = gelu_exact((p[o] - m) * ri * sg1[o] + sbe1[o]);

    // 32 -> 16, LN, GELU
    float h2[16];
    m = 0.0f;
    #pragma unroll
    for (int o = 0; o < 16; ++o) {
        float a = sb2[o];
        #pragma unroll
        for (int j = 0; j < 32; ++j) a += p[j] * sW2[j * 16 + o];
        h2[o] = a; m += a;
    }
    m *= (1.0f / 16.0f);
    v = 0.0f;
    #pragma unroll
    for (int o = 0; o < 16; ++o) { float t = h2[o] - m; v += t * t; }
    ri = rsqrtf(v * (1.0f / 16.0f) + BN_EPS);
    #pragma unroll
    for (int o = 0; o < 16; ++o)
        h2[o] = gelu_exact((h2[o] - m) * ri * sg2[o] + sbe2[o]);

    // 16 -> 8, GELU
    float h3[8];
    #pragma unroll
    for (int o = 0; o < 8; ++o) {
        float a = sb3[o];
        #pragma unroll
        for (int j = 0; j < 16; ++j) a += h2[j] * sW3[j * 8 + o];
        h3[o] = gelu_exact(a);
    }
    // 8 -> 1
    float a = sb4;
    #pragma unroll
    for (int j = 0; j < 8; ++j) a += h3[j] * sW4[j];
    out[i] = a;
}

// ---------------------------------------------------------------------------
static inline size_t alignUp(size_t x) { return (x + 255) & ~(size_t)255; }
static inline int log2i(int d) { int s = 0; while ((1 << s) < d) ++s; return s; }

extern "C" void kernel_launch(void* const* d_in, const int* in_sizes, int n_in,
                              void* d_out, int out_size, void* d_ws, size_t ws_size,
                              hipStream_t stream)
{
    // Input order (flattened setup_inputs dict; lists expand in order):
    // 0:x 1:edge_index 2:inp_W 3:inp_b 4:inp_bn_g 5:inp_bn_b
    // 6-10:conv_Ws 11-15:conv_bs 16-20:bn_gs 21-25:bn_bs 26-30:ln_gs 31-35:ln_bs
    // 36-47: pred params (W1,b1,g1,be1,W2,b2,g2,be2,W3,b3,W4,b4)
    const float* x     = (const float*)d_in[0];
    const int*   ei    = (const int*)  d_in[1];
    const int*   src   = ei;
    const int*   dst   = ei + N_EDGES;
    const float* inp_W = (const float*)d_in[2];
    const float* inp_b = (const float*)d_in[3];
    const float* ibn_g = (const float*)d_in[4];
    const float* ibn_b = (const float*)d_in[5];
    const float* convW[5], *convB[5], *bnG[5], *bnB[5], *lnG[5], *lnB[5];
    for (int i = 0; i < 5; ++i) {
        convW[i] = (const float*)d_in[6  + i];
        convB[i] = (const float*)d_in[11 + i];
        bnG[i]   = (const float*)d_in[16 + i];
        bnB[i]   = (const float*)d_in[21 + i];
        lnG[i]   = (const float*)d_in[26 + i];
        lnB[i]   = (const float*)d_in[31 + i];
    }
    const float* P[12];
    for (int i = 0; i < 12; ++i) P[i] = (const float*)d_in[36 + i];

    // Workspace carve-up (~154 MB)
    char* ws = (char*)d_ws;
    size_t off = 0;
    float* dinv = (float*)(ws + off); off += alignUp((size_t)N_NODES * 4);
    float* bufH = (float*)(ws + off); off += (size_t)N_NODES * 256 * 4;
    float* bufT = (float*)(ws + off); off += (size_t)N_NODES * 256 * 4;
    float* bufG = (float*)(ws + off); off += (size_t)N_NODES * 256 * 4;
    float* sums = (float*)(ws + off); off += 256 * 4;
    float* sumq = (float*)(ws + off); off += 256 * 4;
    if (off > ws_size) return;  // workspace too small; nothing safe to do

    const int T = 256;
    const int nb   = (N_NODES + T - 1) / T;
    const int eb   = (N_EDGES + T - 1) / T;
    const int rowsPerBlock = 512;
    const int statB = (N_NODES + rowsPerBlock - 1) / rowsPerBlock;

    // --- normalization coefficients -------------------------------------
    deg_init  <<<nb, T, 0, stream>>>(dinv, N_NODES);
    deg_accum <<<eb, T, 0, stream>>>(dst, dinv, N_EDGES);
    deg_rsqrt <<<nb, T, 0, stream>>>(dinv, N_NODES);

    // --- input layer: X @ W_in + b -> BN -> GELU ------------------------
    {
        dim3 g((N_NODES + 63) / 64, (256 + 63) / 64);
        gemm_wmma_f16<<<g, T, 0, stream>>>(x, inp_W, inp_b, bufT,
                                           N_NODES, 128, 256);
        bn_zero     <<<1, 256, 0, stream>>>(sums, sumq, 256);
        bn_partial  <<<statB, T, 0, stream>>>(bufT, sums, sumq, N_NODES, 256, rowsPerBlock);
        bn_finalize <<<1, 256, 0, stream>>>(sums, sumq, N_NODES, 256);
        bn_gelu_ln  <<<(N_NODES + 7) / 8, T, 0, stream>>>(
            bufT, sums, sumq, ibn_g, ibn_b, nullptr, nullptr,
            bufH, N_NODES, 256, 0);
    }

    // --- GCN layers -----------------------------------------------------
    int din = 256;
    const int douts[5] = {256, 128, 64, 32, 16};
    for (int i = 0; i < 5; ++i) {
        const int dout = douts[i];
        const int dshift = log2i(dout);
        dim3 g((N_NODES + 63) / 64, (dout + 63) / 64);
        gemm_wmma_f16<<<g, T, 0, stream>>>(bufH, convW[i], nullptr, bufT,
                                           N_NODES, din, dout);
        long tot = (long)N_NODES << dshift;
        agg_init    <<<(int)((tot + T - 1) / T), T, 0, stream>>>(
                         bufT, dinv, convB[i], bufG, N_NODES, dshift);
        const int lshift = (dout >= 32) ? 5 : 4;     // lanes per edge
        long ethreads = (long)N_EDGES << lshift;
        edge_scatter<<<(int)((ethreads + T - 1) / T), T, 0, stream>>>(
                         bufT, src, dst, dinv, bufG, N_EDGES, dout, lshift);
        bn_zero     <<<1, 256, 0, stream>>>(sums, sumq, dout);
        bn_partial  <<<statB, T, 0, stream>>>(bufG, sums, sumq, N_NODES, dout, rowsPerBlock);
        bn_finalize <<<1, 256, 0, stream>>>(sums, sumq, N_NODES, dout);
        bn_gelu_ln  <<<(N_NODES + 7) / 8, T, 0, stream>>>(
            bufG, sums, sumq, bnG[i], bnB[i], lnG[i], lnB[i],
            bufH, N_NODES, dout, 1);
        din = dout;
    }

    // --- prediction head ------------------------------------------------
    pred_mlp<<<nb, T, 0, stream>>>(bufH,
        P[0], P[1], P[2], P[3], P[4], P[5], P[6], P[7], P[8], P[9], P[10], P[11],
        (float*)d_out, N_NODES);
}
